// WindowAttention_36352603193925
// MI455X (gfx1250) — matrix operand order repaired and checked by reading, not compiled, not run on hardware
//
#include <hip/hip_runtime.h>
#include <hip/hip_bf16.h>

typedef __attribute__((ext_vector_type(16))) __bf16    v16bf;
typedef __attribute__((ext_vector_type(16))) _Float16  v16h;
typedef __attribute__((ext_vector_type(8)))  float     v8f;

#define NTOK 49          // window tokens (7x7)
#define CDIM 512
#define HEADS 16
#define HD 32
#define QKVC (3 * CDIM)  // 1536
#define QSCALE 0.17677669529663689f  // 1/sqrt(32)
#define MROWS (2048 * NTOK)          // 100352

// ---------------------------------------------------------------------------
// Prologue: convert x (f32 -> bf16). 4 elems/thread.
// ---------------------------------------------------------------------------
__global__ __launch_bounds__(256) void conv_x_kernel(
    const float* __restrict__ x, __bf16* __restrict__ xbf) {
  const size_t i = ((size_t)blockIdx.x * 256 + threadIdx.x) * 4;
#pragma unroll
  for (int j = 0; j < 4; ++j) xbf[i + j] = (__bf16)x[i + j];
}

// ---------------------------------------------------------------------------
// Prologue: convert qkv_w (f32 -> bf16), folding the q scale (1/sqrt(hd))
// into the first 512 output rows. w is [1536][512] row-major.
// ---------------------------------------------------------------------------
__global__ __launch_bounds__(256) void conv_w_kernel(
    const float* __restrict__ w, __bf16* __restrict__ wbf) {
  const size_t i = ((size_t)blockIdx.x * 256 + threadIdx.x) * 4;
  const float sc = (i < (size_t)CDIM * CDIM) ? QSCALE : 1.0f;  // row < 512
#pragma unroll
  for (int j = 0; j < 4; ++j) wbf[i + j] = (__bf16)(w[i + j] * sc);
}

// ---------------------------------------------------------------------------
// Kernel 1: qkv = x @ w^T + b (bf16 in, f32 accum, bf16 out; q pre-scaled).
// Grid: (100352/16, 1536/512) = (6272, 3), block = 128 (4 waves).
// Wave computes 16(M) x 128(N): 8 accumulators, A fragment reused 8x.
// Inner loop is pure bf16 b128 loads + v_wmma_f32_16x16x32_bf16.
// ---------------------------------------------------------------------------
__global__ __launch_bounds__(128) void qkv_gemm_kernel(
    const __bf16* __restrict__ xbf, const __bf16* __restrict__ wbf,
    const float* __restrict__ bias, __bf16* __restrict__ qkv) {
  const int wave = threadIdx.x >> 5;
  const int lane = threadIdx.x & 31;
  const int lhi  = lane >> 4;      // K-half select
  const int llo  = lane & 15;
  const int m0   = blockIdx.x * 16;
  const int n0   = blockIdx.y * 512 + wave * 128;

  const __bf16* xrow = xbf + ((size_t)(m0 + llo) << 9) + lhi * 8;

  v8f acc[8] = {v8f{}, v8f{}, v8f{}, v8f{}, v8f{}, v8f{}, v8f{}, v8f{}};

  for (int k = 0; k < CDIM; k += 32) {
    // A fragment (16x32): lane<16 -> K {0..7,16..23}; lane>=16 -> {8..15,24..31}
    v16bf a;
    {
      const __bf16* ap = xrow + k;   // already offset by lhi*8
#pragma unroll
      for (int i = 0; i < 8; ++i) {
        a[i]     = ap[i];
        a[8 + i] = ap[16 + i];
      }
    }
    // B fragments (32x16): lane<16 -> K {0..7,16..23}; lane>=16 -> {8..15,24..31}.
    // B[kk][n] = w[n][kk] -> two 8-bf16 (16B) runs of w row n per lane.
#pragma unroll
    for (int t = 0; t < 8; ++t) {
      const __bf16* pw = wbf + ((size_t)(n0 + t * 16 + llo) << 9) + k + lhi * 8;
      v16bf bfrag;
#pragma unroll
      for (int i = 0; i < 8; ++i) {
        bfrag[i]     = pw[i];
        bfrag[8 + i] = pw[16 + i];
      }
      acc[t] = __builtin_amdgcn_wmma_f32_16x16x32_bf16(
          false, a, false, bfrag, (short)0, acc[t], false, false);
    }
  }

  // Store: D layout -> lane = n within tile, VGPR r -> row m0 + r + 8*lhi.
#pragma unroll
  for (int t = 0; t < 8; ++t) {
    const int col  = n0 + t * 16 + llo;
    const float bv = bias[col] * ((col < CDIM) ? QSCALE : 1.0f);
#pragma unroll
    for (int r = 0; r < 8; ++r) {
      const int row = m0 + r + lhi * 8;
      qkv[(size_t)row * QKVC + col] = (__bf16)(acc[t][r] + bv);
    }
  }
}

// ---------------------------------------------------------------------------
// Kernel 2: per-(window, head) attention.
//   S = Qs K^T + rel_pos_bias + mask  (49 padded to 64 with -inf columns)
//   P = softmax(S);  out = P V
// Grid: 2048*16 blocks, block = 128 (4 waves); wave qb owns query rows
// 16qb..16qb+15. bf16 WMMA for S, f16 WMMA for P*V, fp32 softmax.
// V staged transposed in LDS so B fragments are contiguous reads.
// ---------------------------------------------------------------------------
__global__ __launch_bounds__(128) void win_attn_kernel(
    const __bf16* __restrict__ qkv, const float* __restrict__ mask,
    const float* __restrict__ rpb, float* __restrict__ out) {
  const int b    = blockIdx.x >> 4;
  const int h    = blockIdx.x & 15;
  const int tid  = threadIdx.x;
  const int wave = tid >> 5;
  const int lane = tid & 31;
  const int lhi  = lane >> 4;
  const int llo  = lane & 15;

  __shared__ _Float16 Vt[HD][72];         // V transposed [d][key], padded
  __shared__ _Float16 Plds[4][16][72];    // probs, padded stride (bank spread)

  // --- stage V (bf16 -> f16) transposed: 128 threads x 16 elems = 64x32 ---
  {
    const int key = tid >> 1;              // 0..63
    const int c0  = (tid & 1) * 16;
    const __bf16* vp =
        qkv + (size_t)(b * NTOK + key) * QKVC + 2 * CDIM + h * HD + c0;
#pragma unroll
    for (int i = 0; i < 16; ++i)
      Vt[c0 + i][key] = (key < NTOK) ? (_Float16)(float)vp[i] : (_Float16)0.0f;
  }
  __syncthreads();

  // --- Q fragment (A layout, 16x32 bf16) ---
  v16bf aq = {};
  {
    const int qrow = wave * 16 + llo;
    if (qrow < NTOK) {
      const __bf16* qp = qkv + (size_t)(b * NTOK + qrow) * QKVC + h * HD + lhi * 8;
#pragma unroll
      for (int i = 0; i < 8; ++i) {
        aq[i]     = qp[i];
        aq[8 + i] = qp[16 + i];
      }
    }
  }

  // --- S = Qs K^T over 4 key tiles ---
  v8f s[4] = {v8f{}, v8f{}, v8f{}, v8f{}};
#pragma unroll
  for (int kb = 0; kb < 4; ++kb) {
    v16bf bk = {};
    const int krow = kb * 16 + llo;        // B lane = key-tile column index
    if (krow < NTOK) {
      const __bf16* kp =
          qkv + (size_t)(b * NTOK + krow) * QKVC + CDIM + h * HD + lhi * 8;
#pragma unroll
      for (int i = 0; i < 8; ++i) {
        bk[i]     = kp[i];
        bk[8 + i] = kp[16 + i];
      }
    }
    s[kb] = __builtin_amdgcn_wmma_f32_16x16x32_bf16(
        false, aq, false, bk, (short)0, s[kb], false, false);
  }

  // --- bias + mask + softmax (each row lives in one 16-lane half) ---
  const float* mwin = mask + (size_t)(b & 63) * NTOK * NTOK;
#pragma unroll
  for (int r = 0; r < 8; ++r) {
    const int m = wave * 16 + r + lhi * 8;   // query row (uniform across half)
    float vals[4];
#pragma unroll
    for (int kb = 0; kb < 4; ++kb) {
      const int n = kb * 16 + llo;           // key column
      float v = -3.0e30f;
      if (m < NTOK && n < NTOK) {
        const int i1 = m / 7, j1 = m % 7, i2 = n / 7, j2 = n % 7;
        const int ridx = (i1 - i2 + 6) * 13 + (j1 - j2 + 6);
        v = s[kb][r] + rpb[ridx * HEADS + h] + mwin[m * NTOK + n];
      }
      vals[kb] = v;
    }
    float mx = fmaxf(fmaxf(vals[0], vals[1]), fmaxf(vals[2], vals[3]));
#pragma unroll
    for (int d = 1; d < 16; d <<= 1) mx = fmaxf(mx, __shfl_xor(mx, d, 32));
    float sum = 0.0f;
#pragma unroll
    for (int kb = 0; kb < 4; ++kb) {
      vals[kb] = __expf(vals[kb] - mx);
      sum += vals[kb];
    }
#pragma unroll
    for (int d = 1; d < 16; d <<= 1) sum += __shfl_xor(sum, d, 32);
    const float inv = 1.0f / sum;
#pragma unroll
    for (int kb = 0; kb < 4; ++kb)
      Plds[wave][r + lhi * 8][kb * 16 + llo] = (_Float16)(vals[kb] * inv);
  }
  __syncthreads();

  // --- out = P V : A = P (16x32 f16) x2 K-steps, B = V^T rows (32x16) x2 ---
  v8f o[2] = {v8f{}, v8f{}};
#pragma unroll
  for (int kk = 0; kk < 2; ++kk) {
    v16h ap;
    {
      const _Float16* pp = &Plds[wave][llo][kk * 32 + lhi * 8];
#pragma unroll
      for (int i = 0; i < 8; ++i) {
        ap[i]     = pp[i];
        ap[8 + i] = pp[16 + i];
      }
    }
#pragma unroll
    for (int nt = 0; nt < 2; ++nt) {
      // B element (k=key, n=d) = V[key][d] = Vt[d][key]; contiguous in Vt.
      const _Float16* vp = &Vt[nt * 16 + llo][kk * 32 + lhi * 8];
      v16h bv;
#pragma unroll
      for (int i = 0; i < 8; ++i) {
        bv[i]     = vp[i];
        bv[8 + i] = vp[16 + i];
      }
      o[nt] = __builtin_amdgcn_wmma_f32_16x16x32_f16(
          false, ap, false, bv, (short)0, o[nt], false, false);
    }
  }

  // --- store fp32 out[b, m, h*32 + d] ---
#pragma unroll
  for (int nt = 0; nt < 2; ++nt) {
    const int d = nt * 16 + llo;
#pragma unroll
    for (int r = 0; r < 8; ++r) {
      const int m = wave * 16 + r + lhi * 8;
      if (m < NTOK)
        out[(size_t)(b * NTOK + m) * CDIM + h * HD + d] = o[nt][r];
    }
  }
}

extern "C" void kernel_launch(void* const* d_in, const int* in_sizes, int n_in,
                              void* d_out, int out_size, void* d_ws, size_t ws_size,
                              hipStream_t stream) {
  const float* x     = (const float*)d_in[0];  // [2048, 49, 512]
  const float* mask  = (const float*)d_in[1];  // [64, 49, 49]
  const float* qkv_w = (const float*)d_in[2];  // [1536, 512]
  const float* qkv_b = (const float*)d_in[3];  // [1536]
  const float* rpb   = (const float*)d_in[4];  // [169, 16]
  float* out = (float*)d_out;                  // [2048, 49, 512]

  // Workspace layout (bf16): qkv [100352,1536], xbf [100352,512], wbf [1536,512]
  char* ws = (char*)d_ws;
  __bf16* qkv = (__bf16*)ws;                                    // 308,281,344 B
  __bf16* xbf = (__bf16*)(ws + (size_t)MROWS * QKVC * 2);       // 102,760,448 B
  __bf16* wbf = (__bf16*)(ws + (size_t)MROWS * QKVC * 2 +
                               (size_t)MROWS * CDIM * 2);       //   1,572,864 B

  // Prologue conversions (exact multiples of 1024 elements each)
  conv_x_kernel<<<dim3((MROWS * CDIM) / 1024), 256, 0, stream>>>(x, xbf);
  conv_w_kernel<<<dim3((QKVC * CDIM) / 1024), 256, 0, stream>>>(qkv_w, wbf);

  // QKV projection: M=100352 (6272 tiles of 16), N=1536 (3 slabs of 512)
  qkv_gemm_kernel<<<dim3(6272, 3), 128, 0, stream>>>(xbf, wbf, qkv_b, qkv);

  // Attention: one block per (window, head)
  win_attn_kernel<<<dim3(2048 * HEADS), 128, 0, stream>>>(qkv, mask, rpb, out);
}
